// Basic_RSNN_spike2_88845693485074
// MI455X (gfx1250) — matrix operand order repaired and checked by reading, not compiled, not run on hardware
//
#include <hip/hip_runtime.h>
#include <hip/hip_bf16.h>

typedef _Float16 half8  __attribute__((ext_vector_type(8)));
typedef _Float16 half16 __attribute__((ext_vector_type(16)));
typedef float    float8 __attribute__((ext_vector_type(8)));
typedef int      i128v  __attribute__((vector_size(16)));   // 16B payload type

#define N_IN      700
#define N_IN_PAD  704     // pad K to multiple of 32, 16B-aligned rows
#define N_TOT     2068    // 2048 + 20
#define N_PAD     2080    // pad K to multiple of 32, 16B-aligned rows
#define BATCH     256
#define TSTEPS    100
#define NOUT      20
#define TAU       0.6f

#define BM 64
#define BN 128
#define BK 32
#define THREADS 256       // 8 waves (wave32)

// ---- CDNA5 async global->LDS path (ASYNCcnt), guarded so fallback compiles ----
#if defined(__gfx1250__) && __has_builtin(__builtin_amdgcn_global_load_async_to_lds_b128)
#define HAS_ASYNC_LDS 1
#else
#define HAS_ASYNC_LDS 0
#endif

#if HAS_ASYNC_LDS
typedef __attribute__((address_space(1))) i128v* gbl128_t;
typedef __attribute__((address_space(3))) i128v* lds128_t;

__device__ __forceinline__ void async_cp16(const _Float16* g, _Float16* l) {
    __builtin_amdgcn_global_load_async_to_lds_b128(
        (gbl128_t)(void*)g, (lds128_t)(void*)l, 0, 0);
}
__device__ __forceinline__ void wait_async0() {
#if __has_builtin(__builtin_amdgcn_s_wait_asynccnt)
    __builtin_amdgcn_s_wait_asynccnt(0);
#else
    asm volatile("s_wait_asynccnt 0" ::: "memory");
#endif
}
#endif

// ---------------------------------------------------------------- helpers
__device__ __forceinline__ int swz(int row, int chunk) {
    // XOR-swizzle 16B chunks within a 64B LDS row to avoid bank conflicts
    return chunk ^ (row & 3);
}

__device__ __forceinline__ half16 mk16(half8 lo, half8 hi) {
    half16 r;
#pragma unroll
    for (int i = 0; i < 8; ++i) { r[i] = lo[i]; r[i + 8] = hi[i]; }
    return r;
}

// ---------------------------------------------------------------- converts
__global__ void cvt_pad_f16(const float* __restrict__ src, _Float16* __restrict__ dst,
                            int rows, int src_cols, int dst_cols) {
    long idx = (long)blockIdx.x * blockDim.x + threadIdx.x;
    long total = (long)rows * dst_cols;
    if (idx >= total) return;
    int r = (int)(idx / dst_cols);
    int c = (int)(idx % dst_cols);
    dst[idx] = (c < src_cols) ? (_Float16)src[(long)r * src_cols + c] : (_Float16)0.f;
}

__global__ void cvt_transpose_f16(const float* __restrict__ src, _Float16* __restrict__ dst) {
    // src: W_rec [k][m] (N_TOT x N_TOT), dst: W_recT f16 [m][k] stride N_PAD, K zero-padded
    long idx = (long)blockIdx.x * blockDim.x + threadIdx.x;
    long total = (long)N_TOT * N_PAD;
    if (idx >= total) return;
    int m = (int)(idx / N_PAD);
    int k = (int)(idx % N_PAD);
    dst[idx] = (k < N_TOT) ? (_Float16)src[(long)k * N_TOT + m] : (_Float16)0.f;
}

__global__ void zero_u32(unsigned* __restrict__ p, long n) {
    long i = (long)blockIdx.x * blockDim.x + threadIdx.x;
    if (i < n) p[i] = 0u;
}

// ---------------------------------------------------------------- GEMM core
// MODE 0: ff = A(x f16) @ BT(W_fc1 f16)^T   -> store ff[t][b][n] f32
// MODE 1: cur = A(spk f16) @ BT(W_recT)^T   -> fused LIF update epilogue
template <int MODE>
__global__ __launch_bounds__(THREADS)
void rsnn_gemm(const _Float16* __restrict__ A, int lda,
               const _Float16* __restrict__ BT, int ldb, int ksteps,
               const float* __restrict__ ff, float* __restrict__ ffout,
               float* __restrict__ mem, const _Float16* __restrict__ spk_prev,
               _Float16* __restrict__ spk_next, float* __restrict__ out, int t)
{
    __shared__ __align__(16) _Float16 sA[2][BM * BK];   // 2 x 4 KB
    __shared__ __align__(16) _Float16 sB[2][BN * BK];   // 2 x 8 KB (stored as [n][k])

    const int tid  = threadIdx.x;
    const int lane = tid & 31;
    const int wid  = tid >> 5;
    const int wm   = wid >> 2;            // 0..1  (32-row slab)
    const int wn   = wid & 3;             // 0..3  (32-col slab)
    const int m0   = blockIdx.x * BM;
    const int n0   = blockIdx.y * BN;

    // cooperative tile-load coordinates: one b128 for A, two for B per thread.
    // Out-of-range B rows are CLAMPED (not zeroed): row nr only feeds output
    // column n==nr, which the epilogue masks, so garbage there is harmless.
    const int arow = tid >> 2;            // 0..63
    const int achk = tid & 3;             // 16B chunk within 64B row
    const int nr0  = (n0 + arow      < N_TOT) ? (n0 + arow)      : (N_TOT - 1);
    const int nr1  = (n0 + arow + 64 < N_TOT) ? (n0 + arow + 64) : (N_TOT - 1);
    const _Float16* gA  = A  + (long)(m0 + arow) * lda + achk * 8;
    const _Float16* gB0 = BT + (long)nr0 * ldb + achk * 8;
    const _Float16* gB1 = BT + (long)nr1 * ldb + achk * 8;
    const int aoff  = arow * BK + swz(arow, achk) * 8;
    const int boff0 = arow * BK + swz(arow, achk) * 8;
    const int boff1 = (arow + 64) * BK + swz(arow + 64, achk) * 8;

    const float8 fz = {};
    float8 acc[2][2];
#pragma unroll
    for (int i = 0; i < 2; ++i)
#pragma unroll
        for (int j = 0; j < 2; ++j) acc[i][j] = fz;

    // fragment loads from LDS (documented 16-bit A/B VGPR layouts) + 4 WMMA
    auto compute = [&](int pb) {
        half16 af[2], bf[2];
        const int c0 = lane >> 4;         // A: lanes>=16 start at K=8
#pragma unroll
        for (int i = 0; i < 2; ++i) {
            int r = wm * 32 + i * 16 + (lane & 15);
            half8 lo = *(const half8*)&sA[pb][r * BK + swz(r, c0) * 8];       // K=kb..kb+7
            half8 hi = *(const half8*)&sA[pb][r * BK + swz(r, c0 + 2) * 8];   // K=kb+16..+23
            af[i] = mk16(lo, hi);
        }
        const int bc = lane >> 4;         // B: lanes>=16 start at K=8
#pragma unroll
        for (int j = 0; j < 2; ++j) {
            int r = wn * 32 + j * 16 + (lane & 15);
            half8 lo = *(const half8*)&sB[pb][r * BK + swz(r, bc) * 8];
            half8 hi = *(const half8*)&sB[pb][r * BK + swz(r, bc + 2) * 8];
            bf[j] = mk16(lo, hi);
        }
#pragma unroll
        for (int i = 0; i < 2; ++i)
#pragma unroll
            for (int j = 0; j < 2; ++j)
                acc[i][j] = __builtin_amdgcn_wmma_f32_16x16x32_f16(
                    false, af[i], false, bf[j], (short)0, acc[i][j], false, false);
    };

#if HAS_ASYNC_LDS
    // -------- ASYNCcnt-tracked double buffering: copies bypass the VGPR file ----
    async_cp16(gA,  &sA[0][aoff]);
    async_cp16(gB0, &sB[0][boff0]);
    async_cp16(gB1, &sB[0][boff1]);
    wait_async0();
    __syncthreads();

    for (int kk = 0; kk < ksteps; ++kk) {
        const int pb = kk & 1;
        if (kk + 1 < ksteps) {            // issue next-tile DMA before compute
            const int k0 = (kk + 1) * BK;
            const int nb = pb ^ 1;
            async_cp16(gA + k0,  &sA[nb][aoff]);
            async_cp16(gB0 + k0, &sB[nb][boff0]);
            async_cp16(gB1 + k0, &sB[nb][boff1]);
        }
        compute(pb);
        wait_async0();                    // my copies into nb have landed
        __syncthreads();                  // everyone's copies have landed
    }
#else
    // -------- fallback: stage through registers ----
    half8 ra, rb0, rb1;
    ra  = *(const half8*)gA;
    rb0 = *(const half8*)gB0;
    rb1 = *(const half8*)gB1;
    *(half8*)&sA[0][aoff]  = ra;
    *(half8*)&sB[0][boff0] = rb0;
    *(half8*)&sB[0][boff1] = rb1;
    __syncthreads();

    for (int kk = 0; kk < ksteps; ++kk) {
        const int pb = kk & 1;
        if (kk + 1 < ksteps) {
            const int k0 = (kk + 1) * BK;
            ra  = *(const half8*)(gA + k0);
            rb0 = *(const half8*)(gB0 + k0);
            rb1 = *(const half8*)(gB1 + k0);
        }
        compute(pb);
        __syncthreads();
        if (kk + 1 < ksteps) {
            const int nb = pb ^ 1;
            *(half8*)&sA[nb][aoff]  = ra;
            *(half8*)&sB[nb][boff0] = rb0;
            *(half8*)&sB[nb][boff1] = rb1;
        }
        __syncthreads();
    }
#endif

    // -------- epilogue (C/D layout: elem v -> M = v + (lane>=16)*8, N = lane&15) ----
#pragma unroll
    for (int i = 0; i < 2; ++i)
#pragma unroll
        for (int j = 0; j < 2; ++j)
#pragma unroll
            for (int v = 0; v < 8; ++v) {
                int ml = wm * 32 + i * 16 + ((lane >> 4) << 3) + v;
                int nl = wn * 32 + j * 16 + (lane & 15);
                int m = m0 + ml, n = n0 + nl;
                if (n >= N_TOT) continue;
                float val = acc[i][j][v];
                if (MODE == 0) {
                    int b = m / TSTEPS, tt = m % TSTEPS;            // row = b*T + t
                    ffout[((long)tt * BATCH + b) * N_TOT + n] = val;
                } else {
                    long idx = (long)m * N_TOT + n;
                    float curv = val + ff[((long)t * BATCH + m) * N_TOT + n];
                    float mv = mem[idx];
                    float sp = (float)spk_prev[(long)m * N_PAD + n];   // 0 or 1
                    float mn = TAU * mv * (1.f - sp) + curv;           // hard-reset LIF
                    float sn = (mn >= 1.f) ? 1.f : 0.f;
                    mem[idx] = mn;
                    spk_next[(long)m * N_PAD + n] = (_Float16)sn;
                    if (n >= N_TOT - NOUT)
                        out[((long)m * TSTEPS + t) * NOUT + (n - (N_TOT - NOUT))] = sn;
                }
            }
}

// ---------------------------------------------------------------- launch
extern "C" void kernel_launch(void* const* d_in, const int* in_sizes, int n_in,
                              void* d_out, int out_size, void* d_ws, size_t ws_size,
                              hipStream_t stream) {
    const float* x     = (const float*)d_in[0];   // [256,100,700]
    const float* W_fc1 = (const float*)d_in[1];   // [2068,700]
    const float* W_rec = (const float*)d_in[2];   // [2068,2068]
    float* out = (float*)d_out;                   // [256,100,20]

    char* ws = (char*)d_ws;
    auto take = [&](size_t bytes) -> char* {
        char* p = ws;
        ws += (bytes + 255) & ~(size_t)255;
        return p;
    };
    const long MROWS = (long)BATCH * TSTEPS;                       // 25600
    _Float16* x16  = (_Float16*)take((size_t)MROWS * N_IN_PAD * 2);
    _Float16* w116 = (_Float16*)take((size_t)N_TOT * N_IN_PAD * 2);
    _Float16* wrT  = (_Float16*)take((size_t)N_TOT * N_PAD * 2);
    float*    ffb  = (float*)   take((size_t)TSTEPS * BATCH * N_TOT * 4);
    float*    memb = (float*)   take((size_t)BATCH * N_TOT * 4);
    _Float16* spk0 = (_Float16*)take((size_t)BATCH * N_PAD * 2);
    _Float16* spk1 = (_Float16*)take((size_t)BATCH * N_PAD * 2);

    // precision conversions / weight transpose (one-time, HBM-trivial)
    {
        long tot = MROWS * N_IN_PAD;
        cvt_pad_f16<<<(tot + 255) / 256, 256, 0, stream>>>(x, x16, (int)MROWS, N_IN, N_IN_PAD);
    }
    {
        long tot = (long)N_TOT * N_IN_PAD;
        cvt_pad_f16<<<(tot + 255) / 256, 256, 0, stream>>>(W_fc1, w116, N_TOT, N_IN, N_IN_PAD);
    }
    {
        long tot = (long)N_TOT * N_PAD;
        cvt_transpose_f16<<<(tot + 255) / 256, 256, 0, stream>>>(W_rec, wrT);
    }
    {   // zero mem + both spike buffers (incl. K padding columns)
        long nm = (long)BATCH * N_TOT;
        zero_u32<<<(nm + 255) / 256, 256, 0, stream>>>((unsigned*)memb, nm);
        long ns = (long)BATCH * N_PAD / 2;
        zero_u32<<<(ns + 255) / 256, 256, 0, stream>>>((unsigned*)spk0, ns);
        zero_u32<<<(ns + 255) / 256, 256, 0, stream>>>((unsigned*)spk1, ns);
    }

    // phase 1: feed-forward GEMM  (25600 x 2068 x 704)
    dim3 g1(MROWS / BM, (N_TOT + BN - 1) / BN);
    rsnn_gemm<0><<<g1, THREADS, 0, stream>>>(x16, N_IN_PAD, w116, N_IN_PAD, N_IN_PAD / BK,
                                             nullptr, ffb, nullptr, nullptr, nullptr,
                                             nullptr, 0);

    // phase 2: 100 sequential fused (spk @ W_rec + LIF) steps; L2-resident
    dim3 g2(BATCH / BM, (N_TOT + BN - 1) / BN);
    for (int t = 0; t < TSTEPS; ++t) {
        _Float16* sprev = (t & 1) ? spk1 : spk0;
        _Float16* snext = (t & 1) ? spk0 : spk1;
        rsnn_gemm<1><<<g2, THREADS, 0, stream>>>(sprev, N_PAD, wrT, N_PAD, N_PAD / BK,
                                                 ffb, nullptr, memb, sprev, snext,
                                                 out, t);
    }
}
